// ScaledDotProductAttention_69398081569423
// MI455X (gfx1250) — matrix operand order repaired and checked by reading, not compiled, not run on hardware
//
#include <hip/hip_runtime.h>
#include <hip/hip_bf16.h>

namespace {

constexpr int NB = 2, NH = 16, NS = 2048, NDK = 128, NDV = 128;
constexpr int QT  = 16;          // query rows per workgroup
constexpr int LDQ = NDK + 4;     // padded f16 row stride for Q/K LDS tiles (132*2B = 66 banks)
constexpr int LDP = 34;          // padded f16 row stride for P LDS tile

typedef __attribute__((ext_vector_type(16))) _Float16 v16h;
typedef __attribute__((ext_vector_type(8)))  float    v8f;
typedef __attribute__((ext_vector_type(4)))  _Float16 h4;

__device__ __forceinline__ v8f wmma16(v16h a, v16h b, v8f c) {
  // D = A(16x32 f16) * B(32x16 f16) + C(16x16 f32)
  return __builtin_amdgcn_wmma_f32_16x16x32_f16(false, a, false, b, (short)0, c,
                                                false, false);
}

__global__ __launch_bounds__(32)
void sdpa_wmma_kernel(const float* __restrict__ Q,
                      const float* __restrict__ K,
                      const float* __restrict__ V,
                      const unsigned char* __restrict__ amask,
                      float* __restrict__ ctx,
                      float* __restrict__ attn)
{
  __shared__ _Float16 Qs[QT * LDQ];       // 16 x 128 Q tile (f16)
  __shared__ _Float16 Ks[QT * LDQ];       // 16 x 128 K tile (f16)
  __shared__ _Float16 Vs[32 * NDV];       // 32 x 128 V chunk (f16)
  __shared__ _Float16 Ps[QT * LDP];       // 16 x 32  P tile  (f16)

  const int lane = threadIdx.x;           // wave32
  const int ln   = lane & 15;             // col (N) / row (M) index within tile
  const int hi   = lane >> 4;             // which 16-lane half

  const int qblk = blockIdx.x;            // 0..127
  const int bh   = blockIdx.y;            // 0..31
  const int b    = bh >> 4;               // NH == 16

  const int q0 = qblk * QT;
  const float* Qp = Q + (size_t)bh * NS * NDK;
  const float* Kp = K + (size_t)bh * NS * NDK;
  const float* Vp = V + (size_t)bh * NS * NDV;
  float* attnP = attn + ((size_t)bh * NS + q0) * NS;   // 16 x NS block
  float* ctxP  = ctx  + ((size_t)bh * NS + q0) * NDV;  // 16 x NDV block
  const unsigned char* maskP = amask + (size_t)b * NS * NS + (size_t)q0 * NS;

  const float scale = 0.08838834764831845f;            // 1/sqrt(128)

  // ---------------- stage Q tile to LDS as f16 (coalesced b128 loads) --------
  {
    const float4* qg = (const float4*)(Qp + (size_t)q0 * NDK);
    #pragma unroll
    for (int i = 0; i < 16; ++i) {                     // 512 float4 / 32 lanes
      int idx4 = lane + 32 * i;
      int row  = idx4 >> 5;                            // 32 float4 per row
      int c4   = idx4 & 31;
      float4 v = qg[idx4];
      h4 hv; hv[0] = (_Float16)v.x; hv[1] = (_Float16)v.y;
             hv[2] = (_Float16)v.z; hv[3] = (_Float16)v.w;
      *(h4*)&Qs[row * LDQ + c4 * 4] = hv;
    }
  }
  __syncthreads();

  // Persistent A fragments for Q: 4 chunks of K=32
  v16h aQ[4];
  #pragma unroll
  for (int c = 0; c < 4; ++c) {
    #pragma unroll
    for (int h = 0; h < 16; ++h) {
      int k = (h & 7) + 8 * (((h >> 3) << 1) + hi);    // ISA 16-bit A layout
      aQ[c][h] = Qs[ln * LDQ + c * 32 + k];
    }
  }

  // ---------------- Pass A: masked scaled scores + row max -------------------
  float m[8];
  #pragma unroll
  for (int r = 0; r < 8; ++r) m[r] = -3.0e38f;

  for (int kt = 0; kt < NS / 16; ++kt) {
    __syncthreads();
    {
      const float4* kg = (const float4*)(Kp + (size_t)(kt * 16) * NDK);
      #pragma unroll
      for (int i = 0; i < 16; ++i) {
        int idx4 = lane + 32 * i;
        int row  = idx4 >> 5;
        int c4   = idx4 & 31;
        float4 v = kg[idx4];
        h4 hv; hv[0] = (_Float16)v.x; hv[1] = (_Float16)v.y;
               hv[2] = (_Float16)v.z; hv[3] = (_Float16)v.w;
        *(h4*)&Ks[row * LDQ + c4 * 4] = hv;
      }
    }
    __syncthreads();

    v8f acc = {};
    #pragma unroll
    for (int c = 0; c < 4; ++c) {
      v16h bK;                                         // B[k][n] = K[n][k]
      #pragma unroll
      for (int h = 0; h < 16; ++h)
        bK[h] = Ks[ln * LDQ + c * 32 + (h & 7) + 8 * (((h >> 3) << 1) + hi)];   // ISA 16-bit B layout
      acc = wmma16(aQ[c], bK, acc);
    }

    const int jcol = kt * 16 + ln;
    #pragma unroll
    for (int r = 0; r < 8; ++r) {
      int rloc = r + 8 * hi;
      int irow = q0 + rloc;
      float s = acc[r] * scale;
      bool win = jcol < (irow * 4) / 5;                // short-window mask
      bool mm  = maskP[(size_t)rloc * NS + jcol] != 0;
      if (win | mm) s = -1.0e9f;
      attnP[(size_t)rloc * NS + jcol] = s;
      m[r] = fmaxf(m[r], s);
    }
  }
  // row max across the 16-lane half that shares each row
  #pragma unroll
  for (int off = 1; off < 16; off <<= 1) {
    #pragma unroll
    for (int r = 0; r < 8; ++r)
      m[r] = fmaxf(m[r], __shfl_xor(m[r], off, 16));
  }

  // ---------------- Pass B: exp, row sums, O = P @ V -------------------------
  v8f o[8];
  #pragma unroll
  for (int nt = 0; nt < 8; ++nt) o[nt] = (v8f){};
  float t[8];
  #pragma unroll
  for (int r = 0; r < 8; ++r) t[r] = 0.0f;

  for (int p2 = 0; p2 < NS / 32; ++p2) {
    const int k0 = p2 * 32;
    __syncthreads();
    #pragma unroll
    for (int half = 0; half < 2; ++half) {
      #pragma unroll
      for (int r = 0; r < 8; ++r) {
        int rloc = r + 8 * hi;
        size_t off = (size_t)rloc * NS + k0 + half * 16 + ln;
        float s  = attnP[off];
        float pe = __expf(s - m[r]);
        t[r] += pe;
        attnP[off] = pe;                               // unnormalized for now
        Ps[rloc * LDP + half * 16 + ln] = (_Float16)pe;
      }
    }
    {
      const float4* vg = (const float4*)(Vp + (size_t)k0 * NDV);
      #pragma unroll
      for (int i = 0; i < 32; ++i) {                   // 1024 float4 / 32 lanes
        int idx4 = lane + 32 * i;
        float4 v = vg[idx4];
        h4 hv; hv[0] = (_Float16)v.x; hv[1] = (_Float16)v.y;
               hv[2] = (_Float16)v.z; hv[3] = (_Float16)v.w;
        ((h4*)Vs)[idx4] = hv;
      }
    }
    __syncthreads();

    v16h aP;                                           // A = P (16 x 32)
    #pragma unroll
    for (int h = 0; h < 16; ++h) {
      int k = (h & 7) + 8 * (((h >> 3) << 1) + hi);
      aP[h] = Ps[ln * LDP + k];
    }
    #pragma unroll
    for (int nt = 0; nt < 8; ++nt) {
      v16h bV;                                         // B[k][n] = V[k0+k][nt*16+n]
      #pragma unroll
      for (int h = 0; h < 16; ++h)
        bV[h] = Vs[((h & 7) + 8 * (((h >> 3) << 1) + hi)) * NDV + nt * 16 + ln];
      o[nt] = wmma16(aP, bV, o[nt]);
    }
  }

  // row sums across the 16-lane half, then 1/sum
  #pragma unroll
  for (int off = 1; off < 16; off <<= 1) {
    #pragma unroll
    for (int r = 0; r < 8; ++r)
      t[r] += __shfl_xor(t[r], off, 16);
  }
  float inv[8];
  #pragma unroll
  for (int r = 0; r < 8; ++r) inv[r] = 1.0f / t[r];

  // context output
  #pragma unroll
  for (int nt = 0; nt < 8; ++nt) {
    #pragma unroll
    for (int r = 0; r < 8; ++r) {
      int rloc = r + 8 * hi;
      ctxP[(size_t)rloc * NDV + nt * 16 + ln] = o[nt][r] * inv[r];
    }
  }

  // broadcast all 16 row inverses to every lane for the coalesced pass C
  float invAll[16];
  #pragma unroll
  for (int r2 = 0; r2 < 16; ++r2)
    invAll[r2] = __shfl(inv[r2 & 7], 16 * (r2 >> 3), 32);

  // ---------------- Pass C: normalize attn (float4 coalesced RMW) ------------
  #pragma unroll 1
  for (int r2 = 0; r2 < 16; ++r2) {
    float iv = invAll[r2];
    float4* rowp = (float4*)(attnP + (size_t)r2 * NS);
    #pragma unroll
    for (int i = 0; i < 16; ++i) {                     // 512 float4 / 32 lanes
      int idx = lane + 32 * i;
      float4 v = rowp[idx];
      v.x *= iv; v.y *= iv; v.z *= iv; v.w *= iv;
      rowp[idx] = v;
    }
  }
}

} // namespace

extern "C" void kernel_launch(void* const* d_in, const int* in_sizes, int n_in,
                              void* d_out, int out_size, void* d_ws, size_t ws_size,
                              hipStream_t stream) {
  const float* Q = (const float*)d_in[0];
  const float* K = (const float*)d_in[1];
  const float* V = (const float*)d_in[2];
  const unsigned char* amask = (const unsigned char*)d_in[3];   // jnp bool = 1 byte

  float* ctx  = (float*)d_out;                                   // [B,H,S,DV]
  float* attn = ctx + (size_t)NB * NH * NS * NDV;                // [B,H,S,S]

  dim3 grid(NS / QT, NB * NH);
  sdpa_wmma_kernel<<<grid, 32, 0, stream>>>(Q, K, V, amask, ctx, attn);
}
